// TGAE_Encoder_GINE_60206851555362
// MI455X (gfx1250) — matrix operand order repaired and checked by reading, not compiled, not run on hardware
//
#include <hip/hip_runtime.h>
#include <hip/hip_bf16.h>
#include <stdint.h>

// ---------------------------------------------------------------------------
// Problem constants (match reference)
// ---------------------------------------------------------------------------
#define NN     10000
#define EE     320000
#define D_IN   128
#define HH     128
#define LL     3
#define D_CAT  256
#define WHD    256
#define D_OUT  128
#define LN_EPS 1e-5f
#define SLOPE  0.1f

typedef __bf16 bf16_t;
typedef __attribute__((ext_vector_type(16))) __bf16 v16bf;
typedef __attribute__((ext_vector_type(8)))  __bf16 v8bf;
typedef __attribute__((ext_vector_type(8)))  float  v8f;

// ---------------------------------------------------------------------------
// Weight transpose + fp32 -> bf16 convert:  dst[n*K + k] = (bf16) src[k*Nn + n]
// ---------------------------------------------------------------------------
__global__ void conv_w_t(const float* __restrict__ src, bf16_t* __restrict__ dst,
                         int K, int Nn) {
    int total = K * Nn;
    for (int j = blockIdx.x * blockDim.x + threadIdx.x; j < total;
         j += gridDim.x * blockDim.x) {
        int n = j / K;
        int k = j - n * K;
        dst[j] = (bf16_t)src[(size_t)k * Nn + n];
    }
}

// fp32 -> bf16 elementwise
__global__ void f2b(const float* __restrict__ s, bf16_t* __restrict__ d, int n) {
    for (int i = blockIdx.x * blockDim.x + threadIdx.x; i < n;
         i += gridDim.x * blockDim.x)
        d[i] = (bf16_t)s[i];
}

// zero fp32 buffer
__global__ void zero_f(float* __restrict__ p, int n) {
    for (int i = blockIdx.x * blockDim.x + threadIdx.x; i < n;
         i += gridDim.x * blockDim.x)
        p[i] = 0.f;
}

// xc[n, 0:128] = x[n];  xc[n, 128:256] = h[n]  (h has leading dim 4*H = 512)
__global__ void concat_xh(const float* __restrict__ x, const float* __restrict__ h,
                          float* __restrict__ xc, int n_nodes) {
    int total = n_nodes * D_CAT;
    for (int i = blockIdx.x * blockDim.x + threadIdx.x; i < total;
         i += gridDim.x * blockDim.x) {
        int n = i >> 8;          // / 256
        int d = i & 255;
        xc[i] = (d < D_IN) ? x[(size_t)n * D_IN + d]
                           : h[(size_t)n * (4 * HH) + (d - D_IN)];
    }
}

// z_bf = bf16(xc + agg)
__global__ void add_to_bf(const float* __restrict__ xc, const float* __restrict__ agg,
                          bf16_t* __restrict__ z, int n) {
    for (int i = blockIdx.x * blockDim.x + threadIdx.x; i < n;
         i += gridDim.x * blockDim.x)
        z[i] = (bf16_t)(xc[i] + agg[i]);
}

// ---------------------------------------------------------------------------
// Fused edge phase: msg = relu(xc[src] + ea*eW + eB); agg[dst] += msg
// 64 threads per edge, float4 per thread. xc/agg are L2-resident (10 MB each).
// Uses native global_atomic_add_f32 (relaxed, device scope).
// ---------------------------------------------------------------------------
__global__ __launch_bounds__(256)
void edge_msg(const long long* __restrict__ ei, const float* __restrict__ ea,
              const float* __restrict__ xc, const float* __restrict__ eWl,
              const float* __restrict__ eBl, float* __restrict__ agg) {
    int t = blockIdx.x * blockDim.x + threadIdx.x;
    int e = t >> 6;                 // 64 threads per edge
    if (e >= EE) return;
    int q = (t & 63) * 4;           // feature offset (0..252)
    long long s = ei[e];
    long long d = ei[EE + e];
    float a = ea[e];

    const float4 xv = *(const float4*)(xc + (size_t)s * D_CAT + q);
    const float4 w  = *(const float4*)(eWl + q);
    const float4 bb = *(const float4*)(eBl + q);

    float m0 = fmaxf(xv.x + a * w.x + bb.x, 0.f);
    float m1 = fmaxf(xv.y + a * w.y + bb.y, 0.f);
    float m2 = fmaxf(xv.z + a * w.z + bb.z, 0.f);
    float m3 = fmaxf(xv.w + a * w.w + bb.w, 0.f);

    float* dp = agg + (size_t)d * D_CAT + q;
    __hip_atomic_fetch_add(dp + 0, m0, __ATOMIC_RELAXED, __HIP_MEMORY_SCOPE_AGENT);
    __hip_atomic_fetch_add(dp + 1, m1, __ATOMIC_RELAXED, __HIP_MEMORY_SCOPE_AGENT);
    __hip_atomic_fetch_add(dp + 2, m2, __ATOMIC_RELAXED, __HIP_MEMORY_SCOPE_AGENT);
    __hip_atomic_fetch_add(dp + 3, m3, __ATOMIC_RELAXED, __HIP_MEMORY_SCOPE_AGENT);
}

// ---------------------------------------------------------------------------
// WMMA GEMM: out[M,NOUT] = A[M,K](bf16) @ Bt[NOUT,K]^T(bf16) + bias
// One wave computes a 16x64 strip: A fragment loaded once per K-step and
// reused across 4 accumulators (4x v_wmma_f32_16x16x32_bf16 per K-step).
// Optional leaky epilogue; writes fp32 (ldf) and/or bf16 (ldob) outputs.
// ---------------------------------------------------------------------------
__global__ __launch_bounds__(128)
void gemm_bf16_wmma(const bf16_t* __restrict__ A, int lda,
                    const bf16_t* __restrict__ Bt,
                    const float* __restrict__ bias,
                    float* __restrict__ outF, int ldf,
                    bf16_t* __restrict__ outB, int ldob,
                    int M, int K, int NOUT, int leaky) {
    const int wave = threadIdx.x >> 5;
    const int lane = threadIdx.x & 31;
    const int hi   = lane >> 4;       // half-wave select
    const int r16  = lane & 15;
    const int m0 = (blockIdx.x * 4 + wave) * 16;   // one m-tile per wave
    const int n0 = blockIdx.y * 64;                // 64-column strip
    if (m0 >= M) return;              // wave-uniform; EXEC stays all-ones

    v8f acc[4] = {v8f{}, v8f{}, v8f{}, v8f{}};
    // A fragment rows: lane holds row (m0+r16); elems 0-7 -> K=k0+8*hi+0..7,
    // elems 8-15 -> K=k0+8*hi+16..23  (ISA 16-bit A 16x32 layout)
    const bf16_t* arow = A + (size_t)(m0 + r16) * lda + hi * 8;
    // B fragments (Bt is [NOUT,K]): lane holds col (n0+j*16+r16); elems 0-7 ->
    // K=k0+8*hi+0..7, elems 8-15 -> K=k0+8*hi+16..23 (same order as A)
    const bf16_t* brow = Bt + (size_t)(n0 + r16) * K + hi * 8;
    const size_t bstep = (size_t)16 * K;

    for (int k0 = 0; k0 < K; k0 += 32) {
        v8bf alo = *(const v8bf*)(arow + k0);
        v8bf ahi = *(const v8bf*)(arow + k0 + 16);
        v16bf a;
#pragma unroll
        for (int i = 0; i < 8; ++i) { a[i] = alo[i]; a[i + 8] = ahi[i]; }
#pragma unroll
        for (int j = 0; j < 4; ++j) {
            v8bf blo = *(const v8bf*)(brow + (size_t)j * bstep + k0);
            v8bf bhi = *(const v8bf*)(brow + (size_t)j * bstep + k0 + 16);
            v16bf b;
#pragma unroll
            for (int i = 0; i < 8; ++i) { b[i] = blo[i]; b[i + 8] = bhi[i]; }
            acc[j] = __builtin_amdgcn_wmma_f32_16x16x32_bf16(
                /*neg_a=*/false, a, /*neg_b=*/false, b,
                /*c_mod=*/(short)0, acc[j], /*reuse_a=*/false, /*reuse_b=*/false);
        }
    }

#pragma unroll
    for (int j = 0; j < 4; ++j) {
        const int col = n0 + j * 16 + r16;
        const float bv = bias ? bias[col] : 0.f;
#pragma unroll
        for (int r = 0; r < 8; ++r) {
            int row = m0 + hi * 8 + r;     // ISA 32-bit C/D 16x16 layout
            float v = acc[j][r] + bv;
            if (leaky) v = (v > 0.f) ? v : SLOPE * v;
            if (outF) outF[(size_t)row * ldf + col] = v;
            if (outB) outB[(size_t)row * ldob + col] = (bf16_t)v;
        }
    }
}

// ---------------------------------------------------------------------------
// Fused LayerNorm(256) + LeakyReLU -> bf16.  One 256-thread block per node.
// ---------------------------------------------------------------------------
__global__ __launch_bounds__(256)
void ln_leaky_bf16(const float* __restrict__ t, const float* __restrict__ g,
                   const float* __restrict__ b, bf16_t* __restrict__ out) {
    __shared__ float r1[8], r2[8], tot[2];
    const int n = blockIdx.x;
    const int d = threadIdx.x;                 // 0..255
    const float v = t[(size_t)n * WHD + d];

    float s1 = v, s2 = v * v;
#pragma unroll
    for (int off = 16; off; off >>= 1) {
        s1 += __shfl_down(s1, off, 32);
        s2 += __shfl_down(s2, off, 32);
    }
    const int wid = d >> 5, lid = d & 31;
    if (lid == 0) { r1[wid] = s1; r2[wid] = s2; }
    __syncthreads();
    if (d == 0) {
        float a1 = 0.f, a2 = 0.f;
#pragma unroll
        for (int i = 0; i < 8; ++i) { a1 += r1[i]; a2 += r2[i]; }
        tot[0] = a1; tot[1] = a2;
    }
    __syncthreads();
    const float mu  = tot[0] * (1.f / WHD);
    const float var = tot[1] * (1.f / WHD) - mu * mu;
    const float rs  = rsqrtf(var + LN_EPS);
    float y = (v - mu) * rs * g[d] + b[d];
    y = (y > 0.f) ? y : SLOPE * y;
    out[(size_t)n * WHD + d] = (bf16_t)y;
}

// ---------------------------------------------------------------------------
// Launch sequence
// ---------------------------------------------------------------------------
extern "C" void kernel_launch(void* const* d_in, const int* in_sizes, int n_in,
                              void* d_out, int out_size, void* d_ws, size_t ws_size,
                              hipStream_t stream) {
    const float*     x    = (const float*)d_in[0];
    const long long* ei   = (const long long*)d_in[1];  // int64 edge_index [2,E]
    const float*     ea   = (const float*)d_in[2];
    const float*     in_W = (const float*)d_in[3];
    const float*     in_b = (const float*)d_in[4];
    const float*     eW   = (const float*)d_in[5];
    const float*     eB   = (const float*)d_in[6];
    const float*     W1   = (const float*)d_in[7];
    const float*     b1   = (const float*)d_in[8];
    const float*     ln_g = (const float*)d_in[9];
    const float*     ln_b = (const float*)d_in[10];
    const float*     W2   = (const float*)d_in[11];
    const float*     b2   = (const float*)d_in[12];
    const float*     W3   = (const float*)d_in[13];
    const float*     b3   = (const float*)d_in[14];
    const float*     outW = (const float*)d_in[15];
    const float*     outb = (const float*)d_in[16];
    float* out = (float*)d_out;

    // ---- workspace bump allocator (256-B aligned) ----
    char* base = (char*)d_ws;
    size_t off = 0;
    auto alloc = [&](size_t bytes) -> char* {
        char* p = base + off;
        off += (bytes + 255) & ~(size_t)255;
        return p;
    };
    bf16_t* in_Wt  = (bf16_t*)alloc((size_t)HH * D_IN * 2);          // [128,128]
    bf16_t* W1t    = (bf16_t*)alloc((size_t)LL * WHD * D_CAT * 2);   // [L][256,256]
    bf16_t* W2t    = (bf16_t*)alloc((size_t)LL * WHD * WHD * 2);
    bf16_t* W3t    = (bf16_t*)alloc((size_t)LL * HH * WHD * 2);      // [L][128,256]
    bf16_t* outWt  = (bf16_t*)alloc((size_t)D_OUT * 4 * HH * 2);     // [128,512]
    bf16_t* x_bf   = (bf16_t*)alloc((size_t)NN * D_IN * 2);
    float*  hidden = (float*) alloc((size_t)NN * 4 * HH * 4);        // [N,512]
    bf16_t* hid_bf = (bf16_t*)alloc((size_t)NN * 4 * HH * 2);
    float*  xc     = (float*) alloc((size_t)NN * D_CAT * 4);
    float*  agg    = (float*) alloc((size_t)NN * D_CAT * 4);
    bf16_t* z_bf   = (bf16_t*)alloc((size_t)NN * D_CAT * 2);
    float*  t1     = (float*) alloc((size_t)NN * WHD * 4);
    bf16_t* t1_bf  = (bf16_t*)alloc((size_t)NN * WHD * 2);
    bf16_t* t2_bf  = (bf16_t*)alloc((size_t)NN * WHD * 2);
    (void)ws_size; (void)n_in; (void)in_sizes; (void)out_size;

    const int TB = 256;
    auto blks = [](int n, int b) { return (n + b - 1) / b; };
    const int MT  = NN / 16;            // 625 m-tiles (exact)
    const int GMX = (MT + 3) / 4;       // 4 waves (m-tiles) per block

    // ---- weights -> bf16, transposed to [Nout, K] ----
    conv_w_t<<<blks(D_IN * HH, TB), TB, 0, stream>>>(in_W, in_Wt, D_IN, HH);
    for (int l = 0; l < LL; ++l) {
        conv_w_t<<<blks(D_CAT * WHD, TB), TB, 0, stream>>>(
            W1 + (size_t)l * D_CAT * WHD, W1t + (size_t)l * WHD * D_CAT, D_CAT, WHD);
        conv_w_t<<<blks(WHD * WHD, TB), TB, 0, stream>>>(
            W2 + (size_t)l * WHD * WHD, W2t + (size_t)l * WHD * WHD, WHD, WHD);
        conv_w_t<<<blks(WHD * HH, TB), TB, 0, stream>>>(
            W3 + (size_t)l * WHD * HH, W3t + (size_t)l * HH * WHD, WHD, HH);
    }
    conv_w_t<<<blks(4 * HH * D_OUT, TB), TB, 0, stream>>>(outW, outWt, 4 * HH, D_OUT);

    // ---- h0 = x @ in_W + in_b  -> hidden[:, 0:128] ----
    f2b<<<blks(NN * D_IN, TB), TB, 0, stream>>>(x, x_bf, NN * D_IN);
    gemm_bf16_wmma<<<dim3(GMX, HH / 64), 128, 0, stream>>>(
        x_bf, D_IN, in_Wt, in_b, hidden, 4 * HH, nullptr, 0, NN, D_IN, HH, 0);

    // ---- GINE layers ----
    for (int l = 0; l < LL; ++l) {
        concat_xh<<<blks(NN * D_CAT, TB), TB, 0, stream>>>(
            x, hidden + (size_t)l * HH, xc, NN);
        zero_f<<<blks(NN * D_CAT, TB), TB, 0, stream>>>(agg, NN * D_CAT);
        edge_msg<<<blks(EE * 64, TB), TB, 0, stream>>>(
            ei, ea, xc, eW + (size_t)l * D_CAT, eB + (size_t)l * D_CAT, agg);
        add_to_bf<<<blks(NN * D_CAT, TB), TB, 0, stream>>>(xc, agg, z_bf, NN * D_CAT);

        // t1 = z @ W1 + b1
        gemm_bf16_wmma<<<dim3(GMX, WHD / 64), 128, 0, stream>>>(
            z_bf, D_CAT, W1t + (size_t)l * WHD * D_CAT, b1 + (size_t)l * WHD,
            t1, WHD, nullptr, 0, NN, D_CAT, WHD, 0);
        // LayerNorm + leaky -> bf16
        ln_leaky_bf16<<<NN, WHD, 0, stream>>>(
            t1, ln_g + (size_t)l * WHD, ln_b + (size_t)l * WHD, t1_bf);
        // t2 = leaky(t1 @ W2 + b2) -> bf16
        gemm_bf16_wmma<<<dim3(GMX, WHD / 64), 128, 0, stream>>>(
            t1_bf, WHD, W2t + (size_t)l * WHD * WHD, b2 + (size_t)l * WHD,
            nullptr, 0, t2_bf, WHD, NN, WHD, WHD, 1);
        // h_{l+1} = t2 @ W3 + b3 -> hidden[:, 128*(l+1) : 128*(l+2)]
        gemm_bf16_wmma<<<dim3(GMX, HH / 64), 128, 0, stream>>>(
            t2_bf, WHD, W3t + (size_t)l * HH * WHD, b3 + (size_t)l * HH,
            hidden + (size_t)(l + 1) * HH, 4 * HH, nullptr, 0, NN, WHD, HH, 0);
    }

    // ---- readout: out = hidden @ out_W + out_b ----
    f2b<<<blks(NN * 4 * HH, TB), TB, 0, stream>>>(hidden, hid_bf, NN * 4 * HH);
    gemm_bf16_wmma<<<dim3(GMX, D_OUT / 64), 128, 0, stream>>>(
        hid_bf, 4 * HH, outWt, outb, out, D_OUT, nullptr, 0, NN, 4 * HH, D_OUT, 0);
}
